// THTN_13185549598956
// MI455X (gfx1250) — matrix operand order repaired and checked by reading, not compiled, not run on hardware
//
#include <hip/hip_runtime.h>
#include <hip/hip_bf16.h>

// ---------------- problem constants (match reference) ----------------
#define NN   40000   // nodes
#define EE   8000    // hyperedges
#define DEG_E 32     // edge degree (De)
#define DEG_V 8      // node degree (Dv)
#define DD   256     // feature dim D
#define QQ   256     // query dim Q
#define EIG  64
#define NC   40
#define LN_EPS 1e-5f

typedef __attribute__((ext_vector_type(16))) __bf16 v16bf;
typedef __attribute__((ext_vector_type(2)))  __bf16 v2bf;
typedef __attribute__((ext_vector_type(8)))  float  v8f;
typedef __attribute__((ext_vector_type(2)))  float  v2f;

// pack two floats into two bf16 in one dword -> single v_cvt_pk_bf16_f32
static __device__ __forceinline__ unsigned pack_bf16x2(float lo, float hi) {
    v2f f; f.x = lo; f.y = hi;
    v2bf b = __builtin_convertvector(f, v2bf);
    return __builtin_bit_cast(unsigned, b);
}

// =====================================================================
// GEMM: C[M,NOUT] = act(A[M,K] @ W[K,NOUT] + bias [+ C_prev])
// Block = 256 threads (8 waves). All waves share one 64-wide N strip
// (nt64); wave w owns M tile mt = mtblk*8 + w -> block computes 128x64.
// W[32x64] k-strips are staged into a DOUBLE-BUFFERED LDS tile as bf16,
// pre-swizzled into the WMMA B-fragment layout [nt][lane][16]; staging of
// step i+1 overlaps WMMA compute of step i (one barrier per step). Each
// lane preloads all four B fragments (8x ds_load_b128), then issues the
// four v_wmma_f32_16x16x32_bf16 back-to-back behind one s_wait_dscnt.
// NOUT is compile-time: staging addresses reduce to saddr+voffset+imm,
// and for NOUT==256 the column guard vanishes (no exec divergence).
// Fragment layouts (wave32, v_wmma_f32_16x16x32_bf16):
//   A 16x32: lane<16 -> row=lane, elems {0..7}=K{0..7}, {8..15}=K{16..23};
//            lane>=16 -> same row, K offset +8.
//   B 32x16: col = lane&15; lanes 0-15 hold K {0..7,16..23}, lanes 16-31 K {8..15,24..31}.
//   C/D:     col = lane&15, row(elem i) = i + 8*(lane>>4).
// =====================================================================
template<int ACT, bool ACC, int NOUT>
__global__ __launch_bounds__(256)
void gemm_wmma_kernel(const float* __restrict__ A, const float* __restrict__ W,
                      const float* __restrict__ bias, float* __restrict__ C,
                      int M, int K)
{
    __shared__ __bf16 lds_b[2][2048];         // 2 x [4 nt][32 lane][16] = 8 KB

    constexpr int NT64 = (NOUT + 63) >> 6;    // N strips
    const int t    = threadIdx.x;
    const int lane = t & 31;
    const int wave = t >> 5;
    const int mtiles = M >> 4;
    const int mtblk = blockIdx.x / NT64;
    const int nt64  = blockIdx.x % NT64;
    int mt = mtblk * 8 + wave;
    const bool mvalid = (mt < mtiles);        // wave-uniform
    if (!mvalid) mt = 0;                      // clamp; stores masked later

    const int l15  = lane & 15;
    const int half = lane >> 4;               // 0 or 1
    const int row  = (mt << 4) + l15;
    const int akoff = half * 8;

    // staging coordinates for this thread (64 cols x 4 kp-groups)
    const int scol = t & 63;                  // 0..63 within the N strip
    const int colg = (nt64 << 6) + scol;      // global column staged
    const bool colok = (NOUT % 64 == 0) ? true : (colg < NOUT);
    const int kgrp = t >> 6;                  // 0..3
    const int snt  = scol >> 4;
    const int sl15 = scol & 15;

    // stage one 32xK strip of W into LDS buffer `buf` (bf16 frag layout)
    auto stage = [&](int k0, int buf) {
        const float* wbase = W + (size_t)k0 * NOUT + colg;
        #pragma unroll
        for (int s = 0; s < 4; ++s) {
            const int kp = kgrp + (s << 2);       // 0..15, all covered
            const int k  = kp << 1;               // even k in 0..30
            float w0 = 0.0f, w1 = 0.0f;
            if (colok) {
                w0 = wbase[(size_t)k * NOUT];         // imm-offset folds
                w1 = wbase[(size_t)(k + 1) * NOUT];
            }
            const int ln = sl15 | (((k >> 3) & 1) << 4);
            const int j  = (k & 7) | ((k >> 4) << 3); // even -> b32-aligned
            *(unsigned*)&lds_b[buf][(((snt << 5) + ln) << 4) + j] = pack_bf16x2(w0, w1);
        }
    };

    v8f acc[4] = {};
    const int nsteps = K >> 5;

    stage(0, 0);
    __syncthreads();

    for (int i = 0; i < nsteps; ++i) {
        const int k0  = i << 5;
        const int cur = i & 1;
        if (i + 1 < nsteps) stage(k0 + 32, cur ^ 1);   // overlaps compute

        // ---- A fragment: 4 aligned float4 loads from lane-private row ----
        const float* arow = A + (size_t)row * K + k0 + akoff;
        __builtin_prefetch(arow + 32, 0, 3);           // speculative: safe OOB
        float4 a0 = *(const float4*)(arow);
        float4 a1 = *(const float4*)(arow + 4);
        float4 a2 = *(const float4*)(arow + 16);
        float4 a3 = *(const float4*)(arow + 20);
        union { unsigned u[8]; v16bf v; } af;
        af.u[0] = pack_bf16x2(a0.x, a0.y);
        af.u[1] = pack_bf16x2(a0.z, a0.w);
        af.u[2] = pack_bf16x2(a1.x, a1.y);
        af.u[3] = pack_bf16x2(a1.z, a1.w);
        af.u[4] = pack_bf16x2(a2.x, a2.y);
        af.u[5] = pack_bf16x2(a2.z, a2.w);
        af.u[6] = pack_bf16x2(a3.x, a3.y);
        af.u[7] = pack_bf16x2(a3.z, a3.w);

        // ---- preload all four B fragments, then 4 WMMAs back-to-back ----
        const __bf16* lb = &lds_b[cur][lane << 4];
        v16bf bf0 = *(const v16bf*)(lb);
        v16bf bf1 = *(const v16bf*)(lb + (32 << 4));
        v16bf bf2 = *(const v16bf*)(lb + (64 << 4));
        v16bf bf3 = *(const v16bf*)(lb + (96 << 4));
        acc[0] = __builtin_amdgcn_wmma_f32_16x16x32_bf16(false, af.v, false, bf0, (short)0, acc[0], false, false);
        acc[1] = __builtin_amdgcn_wmma_f32_16x16x32_bf16(false, af.v, false, bf1, (short)0, acc[1], false, false);
        acc[2] = __builtin_amdgcn_wmma_f32_16x16x32_bf16(false, af.v, false, bf2, (short)0, acc[2], false, false);
        acc[3] = __builtin_amdgcn_wmma_f32_16x16x32_bf16(false, af.v, false, bf3, (short)0, acc[3], false, false);

        __syncthreads();   // next-buffer stores done block-wide before reuse
    }

    // ---- epilogue: bias (+prev) (+relu), masked for M/N tails ----
    if (mvalid) {
        #pragma unroll
        for (int nt = 0; nt < 4; ++nt) {
            const int col = (nt64 << 6) + (nt << 4) + l15;
            if (NOUT % 64 != 0 && col >= NOUT) continue;
            const float bv = bias ? bias[col] : 0.0f;
            #pragma unroll
            for (int i = 0; i < 8; ++i) {
                const int r = (mt << 4) + (half << 3) + i;
                const size_t off = (size_t)r * NOUT + col;
                float v = acc[nt][i] + bv;
                if (ACC) v += C[off];
                if (ACT == 1) v = v > 0.0f ? v : 0.0f;
                C[off] = v;
            }
        }
    }
}

// =====================================================================
// Gather attention over DEG neighbors:
//   attn[d] = leaky_relu(dot(k[idx[d]], q[row]), .01)*scale + cent[row,d]
//   score   = softmax(attn);  h[row,:] = sum_d score[d] * v[idx[d],:]
// One block (256 threads) per row. Q == D == 256.
// =====================================================================
template<int DEG>
__global__ __launch_bounds__(256)
void attn_gather_kernel(const float* __restrict__ q, const float* __restrict__ k,
                        const float* __restrict__ v, const int* __restrict__ idx,
                        const float* __restrict__ cent, float* __restrict__ h,
                        float scale)
{
    const int row = blockIdx.x;
    const int t   = threadIdx.x;
    __shared__ float red[256];
    __shared__ float sc[DEG];
    __shared__ int   sidx[DEG];

    if (t < DEG) sidx[t] = idx[(size_t)row * DEG + t];
    __syncthreads();

    const int gsz = 256 / DEG;
    const int g = t / gsz, j = t % gsz;
    const float* kr = k + (size_t)sidx[g] * QQ;
    const float* qr = q + (size_t)row * QQ;
    float p = 0.0f;
    for (int c = j; c < QQ; c += gsz) p += kr[c] * qr[c];
    red[t] = p;
    __syncthreads();
    for (int off = gsz >> 1; off > 0; off >>= 1) {
        if (j < off) red[t] += red[t + off];
        __syncthreads();
    }
    if (t < DEG) {
        float a = red[t * gsz];
        a = (a > 0.0f ? a : 0.01f * a) * scale + cent[(size_t)row * DEG + t];
        sc[t] = a;
    }
    __syncthreads();
    if (t == 0) {
        float m = sc[0];
        #pragma unroll
        for (int d = 1; d < DEG; ++d) m = fmaxf(m, sc[d]);
        float s = 0.0f;
        #pragma unroll
        for (int d = 0; d < DEG; ++d) { sc[d] = __expf(sc[d] - m); s += sc[d]; }
        const float inv = 1.0f / s;
        #pragma unroll
        for (int d = 0; d < DEG; ++d) sc[d] *= inv;
    }
    __syncthreads();

    float acc = 0.0f;                         // t == channel (D == 256)
    #pragma unroll
    for (int d = 0; d < DEG; ++d)
        acc += sc[d] * v[(size_t)sidx[d] * DD + t];
    h[(size_t)row * DD + t] = acc;
}

// =====================================================================
// out[row,:] = layernorm(a[row,:] + b[row,:]) * g + beta   (D == 256)
// =====================================================================
__global__ __launch_bounds__(256)
void add_layernorm_kernel(const float* __restrict__ a, const float* __restrict__ b,
                          const float* __restrict__ g, const float* __restrict__ beta,
                          float* __restrict__ out)
{
    const int row = blockIdx.x, t = threadIdx.x;
    __shared__ float red[256];
    __shared__ float s_mu, s_rstd;
    const size_t o = (size_t)row * DD + t;
    const float x = a[o] + b[o];
    red[t] = x;
    __syncthreads();
    for (int off = 128; off > 0; off >>= 1) { if (t < off) red[t] += red[t + off]; __syncthreads(); }
    if (t == 0) s_mu = red[0] * (1.0f / DD);
    __syncthreads();
    const float xm = x - s_mu;
    red[t] = xm * xm;
    __syncthreads();
    for (int off = 128; off > 0; off >>= 1) { if (t < off) red[t] += red[t + off]; __syncthreads(); }
    if (t == 0) s_rstd = rsqrtf(red[0] * (1.0f / DD) + LN_EPS);
    __syncthreads();
    out[o] = xm * s_rstd * g[t] + beta[t];
}

// feat_v = t0 + gcn + cs_emb[cidx] + un_emb[uidx]
__global__ __launch_bounds__(256)
void assemble_v_kernel(const float* __restrict__ t0, const float* __restrict__ gcn,
                       const float* __restrict__ cs_emb, const float* __restrict__ un_emb,
                       const int* __restrict__ cidx, const int* __restrict__ uidx,
                       float* __restrict__ out)
{
    const int row = blockIdx.x, t = threadIdx.x;
    const size_t o = (size_t)row * DD + t;
    out[o] = t0[o] + gcn[o]
           + cs_emb[(size_t)cidx[row] * DD + t]
           + un_emb[(size_t)uidx[row] * DD + t];
}

// ---------------- host-side dispatch helper ----------------
static void launch_gemm(const float* A, const float* W, const float* bias, float* C,
                        int M, int K, int Nout, int act, bool acc, hipStream_t s)
{
    const int mtiles = M >> 4;
    const int mtblks = (mtiles + 7) >> 3;
    const int nt     = (Nout + 63) >> 6;
    dim3 g(mtblks * nt), b(256);
    if (Nout == 256) {
        if (acc)           gemm_wmma_kernel<0, true , 256><<<g, b, 0, s>>>(A, W, bias, C, M, K);
        else if (act == 1) gemm_wmma_kernel<1, false, 256><<<g, b, 0, s>>>(A, W, bias, C, M, K);
        else               gemm_wmma_kernel<0, false, 256><<<g, b, 0, s>>>(A, W, bias, C, M, K);
    } else { // classifier head, Nout == 40
        gemm_wmma_kernel<0, false, NC><<<g, b, 0, s>>>(A, W, bias, C, M, K);
    }
}

extern "C" void kernel_launch(void* const* d_in, const int* in_sizes, int n_in,
                              void* d_out, int out_size, void* d_ws, size_t ws_size,
                              hipStream_t stream)
{
    // ---- inputs (setup_inputs dict order) ----
    const float* vfeat      = (const float*)d_in[0];
    const float* efeat      = (const float*)d_in[1];
    const float* eign_vec   = (const float*)d_in[2];
    const float* feat_v_gcn = (const float*)d_in[3];
    const float* cent1      = (const float*)d_in[4];
    const float* cent2      = (const float*)d_in[5];
    const int*   cent_idx   = (const int*)d_in[6];
    const int*   uniq_idx   = (const int*)d_in[7];
    const int*   edge2node  = (const int*)d_in[8];
    const int*   node2edge  = (const int*)d_in[9];
    const float* W_vtx = (const float*)d_in[10]; const float* b_vtx = (const float*)d_in[11];
    const float* W_pe  = (const float*)d_in[12]; const float* b_pe  = (const float*)d_in[13];
    const float* cs_emb= (const float*)d_in[14]; const float* un_emb= (const float*)d_in[15];
    const float* W_kv  = (const float*)d_in[16]; const float* b_kv  = (const float*)d_in[17];
    const float* W_vv  = (const float*)d_in[18]; const float* b_vv  = (const float*)d_in[19];
    const float* W_qe  = (const float*)d_in[20]; const float* b_qe  = (const float*)d_in[21];
    const float* W_ke  = (const float*)d_in[22]; const float* b_ke  = (const float*)d_in[23];
    const float* W_ve  = (const float*)d_in[24]; const float* b_ve  = (const float*)d_in[25];
    const float* W_qv  = (const float*)d_in[26]; const float* b_qv  = (const float*)d_in[27];
    const float* W_l1  = (const float*)d_in[28]; const float* b_l1  = (const float*)d_in[29];
    const float* W_l2  = (const float*)d_in[30]; const float* b_l2  = (const float*)d_in[31];
    const float* W_l3  = (const float*)d_in[32]; const float* b_l3  = (const float*)d_in[33];
    const float* W_l4  = (const float*)d_in[34]; const float* b_l4  = (const float*)d_in[35];
    const float* ln1_g = (const float*)d_in[36]; const float* ln1_b = (const float*)d_in[37];
    const float* ln2_g = (const float*)d_in[38]; const float* ln2_b = (const float*)d_in[39];
    const float* W_cls = (const float*)d_in[40]; const float* b_cls = (const float*)d_in[41];
    float* out = (float*)d_out;

    // ---- workspace carve-up (fp32), with buffer reuse ----
    const size_t ND = (size_t)NN * DD;   // == N*Q
    const size_t ED = (size_t)EE * DD;   // == E*Q
    float* p = (float*)d_ws;
    float* feat_v  = p; p += ND;   // assembled node features (live to stage 2)
    float* buf_nq  = p; p += ND;   // k_n -> q_v -> node FFN tmp
    float* buf_nd  = p; p += ND;   // v_n -> h_v -> f_v
    float* x_v     = p; p += ND;   // gemm temp early, x_v late
    float* e_q     = p; p += ED;   // q_e -> k_e
    float* e_d1    = p; p += ED;   // h_e -> f_e -> v_e
    float* x_e     = p; p += ED;
    float* e_q2    = p; p += ED;   // edge FFN tmp
    float* feat_e2 = p; p += ED;

    const float scale = 1.0f / 16.0f;  // 1/sqrt(Q)

    // ---- feature assembly ----
    launch_gemm(vfeat, W_vtx, b_vtx, x_v, NN, DD, DD, 0, false, stream);
    launch_gemm(eign_vec, W_pe, b_pe, x_v, NN, EIG, DD, 0, true, stream);
    assemble_v_kernel<<<NN, 256, 0, stream>>>(x_v, feat_v_gcn, cs_emb, un_emb,
                                              cent_idx, uniq_idx, feat_v);

    // ---- stage 1: hyperedges attend over member nodes ----
    launch_gemm(feat_v, W_kv, b_kv, buf_nq, NN, DD, QQ, 0, false, stream);   // k_n
    launch_gemm(feat_v, W_vv, b_vv, buf_nd, NN, DD, DD, 0, false, stream);   // v_n
    launch_gemm(efeat,  W_qe, b_qe, e_q,    EE, DD, QQ, 0, false, stream);   // q_e
    attn_gather_kernel<DEG_E><<<EE, 256, 0, stream>>>(e_q, buf_nq, buf_nd,
                                                      edge2node, cent1, e_d1, scale); // h_e
    add_layernorm_kernel<<<EE, 256, 0, stream>>>(e_d1, efeat, ln1_g, ln1_b, x_e);
    launch_gemm(x_e,  W_l1, b_l1, e_q2, EE, DD, QQ, 1, false, stream);       // relu FFN
    launch_gemm(e_q2, W_l2, b_l2, e_d1, EE, QQ, DD, 0, false, stream);       // f_e
    add_layernorm_kernel<<<EE, 256, 0, stream>>>(e_d1, x_e, ln1_g, ln1_b, feat_e2);

    // ---- stage 2: nodes attend over incident hyperedges ----
    launch_gemm(feat_e2, W_ke, b_ke, e_q,  EE, DD, QQ, 0, false, stream);    // k_e
    launch_gemm(feat_e2, W_ve, b_ve, e_d1, EE, DD, DD, 0, false, stream);    // v_e
    launch_gemm(feat_v,  W_qv, b_qv, buf_nq, NN, DD, QQ, 0, false, stream);  // q_v
    attn_gather_kernel<DEG_V><<<NN, 256, 0, stream>>>(buf_nq, e_q, e_d1,
                                                      node2edge, cent2, buf_nd, scale); // h_v
    add_layernorm_kernel<<<NN, 256, 0, stream>>>(buf_nd, feat_v, ln2_g, ln2_b, x_v);
    launch_gemm(x_v,    W_l3, b_l3, buf_nq, NN, DD, QQ, 1, false, stream);   // relu FFN
    launch_gemm(buf_nq, W_l4, b_l4, buf_nd, NN, QQ, DD, 0, false, stream);   // f_v
    add_layernorm_kernel<<<NN, 256, 0, stream>>>(buf_nd, x_v, ln2_g, ln2_b, feat_v);

    // ---- classifier head (Nout = 40, tail-masked in kernel) ----
    launch_gemm(feat_v, W_cls, b_cls, out, NN, DD, NC, 0, false, stream);

    (void)in_sizes; (void)n_in; (void)out_size; (void)ws_size;
}
